// Caption_20667382628957
// MI455X (gfx1250) — matrix operand order repaired and checked by reading, not compiled, not run on hardware
//
#include <hip/hip_runtime.h>

#define HIDDEN 256
#define VOCABN 30522
#define NHEADS 8
#define DFFN   2048
#define BSZ    8
#define CBCH   2048
#define HW2    400
#define TTOK   25
#define MENC   3200   /* 8*400 */
#define MDEC   200    /* 8*25  */
#define MAXTGT 24
#define ENDID  102

// ---------------- CDNA5 WMMA types ----------------
typedef __attribute__((ext_vector_type(16))) __bf16 v16bf;
typedef __attribute__((ext_vector_type(8)))  float  v8f;

union BF16Frag { v16bf v; unsigned short u[16]; uint4 q[2]; };
union AccFrag  { v8f v; float e[8]; };

__device__ __forceinline__ unsigned short f2bf(float x) {
  union { float f; unsigned int i; } c; c.f = x;
  unsigned int u = c.i;
  return (unsigned short)((u + 0x7FFFu + ((u >> 16) & 1u)) >> 16);
}

// ---------------- TDM availability (arity differs across toolchains) ----------------
#if defined(__has_builtin)
#  if __has_builtin(__builtin_amdgcn_tensor_load_to_lds) && __has_builtin(__builtin_amdgcn_s_wait_tensorcnt)
#    define HAVE_TDM 1
#  else
#    define HAVE_TDM 0
#  endif
#else
#  define HAVE_TDM 0
#endif

typedef __attribute__((ext_vector_type(4))) unsigned int u32x4;
typedef __attribute__((ext_vector_type(8))) int i32x8;
typedef __attribute__((ext_vector_type(4))) int i32x4;

// ---------------- WMMA GEMM ----------------
// C[M,N] = act(A[M,K]_bf16 * Bt[N,K]_bf16 + bias), f32 out (+ optional bf16 shadow Cb).
// One wave computes a 16x64 tile (4 accumulators share one A fragment).
// No predicated loads: OOB rows/cols are pointer-clamped; only stores are guarded.
__global__ __launch_bounds__(256)
void gemm_wmma_kernel(const unsigned short* __restrict__ A, const unsigned short* __restrict__ Bt,
                      const float* __restrict__ bias, float* __restrict__ C,
                      unsigned short* __restrict__ Cb,
                      int M, int N, int K, int relu, int tilesN4, int tilesTotal)
{
  int wave = blockIdx.x * 8 + (threadIdx.x >> 5);
  if (wave >= tilesTotal) return;              // wave-uniform: EXEC stays all-1 for WMMA
  int lane = threadIdx.x & 31;
  int tmi = wave / tilesN4, tni = wave % tilesN4;
  int lr = lane & 15, half = lane >> 4;

  int arow = tmi * 16 + lr;
  if (arow >= M) arow = M - 1;                 // clamp: garbage rows never stored
  const unsigned short* ap = A + (size_t)arow * K;

  const unsigned short* bp[4];
  int bcol[4];
  #pragma unroll
  for (int j = 0; j < 4; j++) {
    int c = tni * 64 + j * 16 + lr;
    bcol[j] = c;
    int cc = (c < N) ? c : (N - 1);            // clamp: garbage cols never stored
    bp[j] = Bt + (size_t)cc * K;
  }

  AccFrag acc[4];
  #pragma unroll
  for (int j = 0; j < 4; j++)
    #pragma unroll
    for (int r = 0; r < 8; r++) acc[j].e[r] = 0.0f;

  for (int k0 = 0; k0 < K; k0 += 32) {
    // A fragment (16-bit A 16x32 layout): half 0 -> K {k0..k0+7, k0+16..k0+23},
    //                                     half 1 -> K {k0+8..k0+15, k0+24..k0+31}
    int kb = k0 + half * 8;
    const uint4* ap4 = reinterpret_cast<const uint4*>(ap + kb);
    BF16Frag a;
    a.q[0] = ap4[0];                            // kb .. kb+7
    a.q[1] = ap4[2];                            // kb+16 .. kb+23

    if (k0 + 256 < K)
      __builtin_prefetch((const void*)(bp[0] + k0 + 256), 0, 3);  // global_prefetch_b8

    #pragma unroll
    for (int j = 0; j < 4; j++) {
      // B fragment (B 32x16): half 0 -> K {k0..k0+7, k0+16..k0+23}, half 1 -> K {k0+8..k0+15, k0+24..k0+31}
      const uint4* bp4 = reinterpret_cast<const uint4*>(bp[j] + k0 + half * 8);
      BF16Frag b;
      b.q[0] = bp4[0];
      b.q[1] = bp4[2];
      acc[j].v = __builtin_amdgcn_wmma_f32_16x16x32_bf16(false, a.v, false, b.v,
                                                         (short)0, acc[j].v, false, false);
    }
  }

  int mbase = tmi * 16 + half * 8;             // C/D layout: lane half selects rows 0..7 / 8..15
  #pragma unroll
  for (int j = 0; j < 4; j++) {
    int cc = (bcol[j] < N) ? bcol[j] : (N - 1);
    float bi = (bias != nullptr) ? bias[cc] : 0.0f;
    if (bcol[j] < N) {
      #pragma unroll
      for (int r = 0; r < 8; r++) {
        int m = mbase + r;
        if (m < M) {
          float v = acc[j].e[r] + bi;
          if (relu) v = fmaxf(v, 0.0f);
          C[(size_t)m * N + bcol[j]] = v;
          if (Cb) Cb[(size_t)m * N + bcol[j]] = f2bf(v);
        }
      }
    }
  }
}

// ---------------- TDM L2-warm kernel ----------------
// DMA the vocab-head weight through L2 in 32KB tiles before the serial decode chain.
#define WARM_CHUNK 16384  /* bf16 elems = 32KB */
__global__ __launch_bounds__(32)
void warm_l2_kernel(const unsigned short* __restrict__ w, unsigned short* __restrict__ sink)
{
  __shared__ unsigned short tile[WARM_CHUNK];
#if HAVE_TDM
  unsigned long long ga = (unsigned long long)(size_t)(w + (size_t)blockIdx.x * WARM_CHUNK);
  // D# group0: count=1 | lds_addr=0 | global_addr | type=2 ("image")
  u32x4 g0 = { 1u, 0u, (unsigned)(ga & 0xFFFFFFFFu),
               (unsigned)(((ga >> 32) & 0x01FFFFFFu) | 0x80000000u) };
  // D# group1: data_size=2B; tensor_dim0=16384, tensor_dim1=1; tile_dim0=16384, tile_dim1=1;
  //            tensor_dim0_stride=16384
  i32x8 g1 = { 0x00010000, (int)0x40000000, 0x00010000, (int)0x40000000,
               1, 0x4000, 0, 0 };
  i32x4 z4 = { 0, 0, 0, 0 };
#if __clang_major__ >= 23
  i32x8 z8 = { 0, 0, 0, 0, 0, 0, 0, 0 };
  __builtin_amdgcn_tensor_load_to_lds(g0, g1, z4, z4, z8, 0);
#else
  __builtin_amdgcn_tensor_load_to_lds(g0, g1, z4, z4, 0);
#endif
  __builtin_amdgcn_s_wait_tensorcnt(0);
#else
  for (int i = threadIdx.x * 64; i < WARM_CHUNK; i += 32 * 64)
    __builtin_prefetch((const void*)(w + (size_t)blockIdx.x * WARM_CHUNK + i), 0, 3);
  tile[threadIdx.x] = 0;
  __syncthreads();
#endif
  if (threadIdx.x == 0) sink[blockIdx.x & 1023] = tile[0];
}

// ---------------- weight conversions ----------------
__global__ void conv_bf16_kernel(const float* __restrict__ W, unsigned short* __restrict__ Wt, int n) {
  int i = blockIdx.x * 256 + threadIdx.x;
  if (i < n) Wt[i] = f2bf(W[i]);
}
// Wt[N][K] = bf16(W[K][N])
__global__ void convt_bf16_kernel(const float* __restrict__ W, unsigned short* __restrict__ Wt, int K, int N) {
  int i = blockIdx.x * 256 + threadIdx.x;
  if (i >= K * N) return;
  int n = i / K, k = i % K;
  Wt[i] = f2bf(W[(size_t)k * N + n]);
}

// ---------------- transposes / elementwise (all emit optional bf16 shadows) ----------------
// srcT_bf16[(b*400+hw)*2048 + c] = bf16(src[b][c][hw])
__global__ void transpose_src_kernel(const float* __restrict__ src, unsigned short* __restrict__ dst) {
  int i = blockIdx.x * 256 + threadIdx.x;          // over 8*400*2048 (out index)
  if (i >= BSZ * HW2 * CBCH) return;
  int c  = i % CBCH;
  int r  = i / CBCH;
  int b  = r / HW2, hw = r % HW2;
  dst[i] = f2bf(src[((size_t)b * CBCH + c) * HW2 + hw]);
}
__global__ void transpose_pos_kernel(const float* __restrict__ pos, float* __restrict__ dst) {
  int i = blockIdx.x * 256 + threadIdx.x;          // over 8*400*256
  if (i >= BSZ * HW2 * HIDDEN) return;
  int c  = i % HIDDEN;
  int r  = i / HIDDEN;
  int b  = r / HW2, hw = r % HW2;
  dst[i] = pos[((size_t)b * HIDDEN + c) * HW2 + hw];
}
__global__ void add_kernel(const float* a, const float* b, float* o, unsigned short* ob, int n) {
  int i = blockIdx.x * 256 + threadIdx.x;
  if (i < n) { float r = a[i] + b[i]; o[i] = r; if (ob) ob[i] = f2bf(r); }
}
// o[r][c] = a[r][c] + pe[r % per][c]   (rows of width 256)
__global__ void add_rows_kernel(const float* a, const float* pe, float* o, unsigned short* ob,
                                int rows, int per) {
  int i = blockIdx.x * 256 + threadIdx.x;
  if (i >= rows * HIDDEN) return;
  int r = i / HIDDEN, c = i % HIDDEN;
  float v = a[i] + pe[(r % per) * HIDDEN + c];
  o[i] = v;
  if (ob) ob[i] = f2bf(v);
}
__global__ void copy_kernel(const float* a, float* o, int n) {
  int i = blockIdx.x * 256 + threadIdx.x;
  if (i < n) o[i] = a[i];
}
__global__ void bool_to_int_kernel(const unsigned char* a, int* o, int n) {
  int i = blockIdx.x * 256 + threadIdx.x;
  if (i < n) o[i] = a[i] ? 1 : 0;
}
__global__ void fill_int_kernel(int* o, int v, int n) {
  int i = blockIdx.x * 256 + threadIdx.x;
  if (i < n) o[i] = v;
}

// ---------------- LayerNorm: out = LN(x (+res)) * s + b ; rows of 256, one wave/row ----------------
__global__ __launch_bounds__(256)
void ln_kernel(const float* __restrict__ x, const float* __restrict__ res,
               const float* __restrict__ s, const float* __restrict__ bta,
               float* __restrict__ out, unsigned short* __restrict__ outb, int rows)
{
  int row  = blockIdx.x * 8 + (threadIdx.x >> 5);
  int lane = threadIdx.x & 31;
  if (row >= rows) return;
  size_t base = (size_t)row * HIDDEN;
  float v[8];
  #pragma unroll
  for (int j = 0; j < 8; j++) {
    int c = lane + j * 32;
    v[j] = x[base + c] + (res ? res[base + c] : 0.0f);
  }
  float su = 0.f;
  #pragma unroll
  for (int j = 0; j < 8; j++) su += v[j];
  #pragma unroll
  for (int o = 16; o > 0; o >>= 1) su += __shfl_xor(su, o, 32);
  float mu = su * (1.0f / 256.0f);
  float var = 0.f;
  #pragma unroll
  for (int j = 0; j < 8; j++) { float d = v[j] - mu; var += d * d; }
  #pragma unroll
  for (int o = 16; o > 0; o >>= 1) var += __shfl_xor(var, o, 32);
  var *= (1.0f / 256.0f);
  float inv = rsqrtf(var + 1e-5f);
  #pragma unroll
  for (int j = 0; j < 8; j++) {
    int c = lane + j * 32;
    float r = (v[j] - mu) * inv * s[c] + bta[c];
    out[base + c] = r;
    if (outb) outb[base + c] = f2bf(r);
  }
}

// ---------------- Attention: one block per (b, h, q); head_dim = 32 ----------------
__global__ __launch_bounds__(128)
void attn_kernel(const float* __restrict__ Q, const float* __restrict__ K,
                 const float* __restrict__ V, float* __restrict__ out,
                 unsigned short* __restrict__ outb,
                 int Lq, int Lk, int causal, const int* __restrict__ kpm)
{
  __shared__ float sc[448];
  __shared__ float qv[32];
  __shared__ float redm[4];
  __shared__ float reds[4];
  int bid = blockIdx.x;
  int q = bid % Lq;
  int h = (bid / Lq) % NHEADS;
  int b = bid / (Lq * NHEADS);
  int t = threadIdx.x;
  if (t < 32) qv[t] = Q[((size_t)(b * Lq + q)) * HIDDEN + h * 32 + t];
  __syncthreads();

  float mx = -3.0e38f;
  for (int k = t; k < Lk; k += 128) {
    const float* kp = K + ((size_t)(b * Lk + k)) * HIDDEN + h * 32;
    float d = 0.f;
    #pragma unroll
    for (int e = 0; e < 32; e++) d += qv[e] * kp[e];
    float sv = d * 0.17677669529663689f;      // 1/sqrt(32)
    if (causal && k > q) sv += -1.0e9f;
    if (kpm && kpm[b * Lk + k]) sv = -1.0e9f;
    sc[k] = sv;
    mx = fmaxf(mx, sv);
  }
  #pragma unroll
  for (int o = 16; o > 0; o >>= 1) mx = fmaxf(mx, __shfl_xor(mx, o, 32));
  if ((t & 31) == 0) redm[t >> 5] = mx;
  __syncthreads();
  mx = fmaxf(fmaxf(redm[0], redm[1]), fmaxf(redm[2], redm[3]));

  float sum = 0.f;
  for (int k = t; k < Lk; k += 128) {
    float e = __expf(sc[k] - mx);
    sc[k] = e;
    sum += e;
  }
  #pragma unroll
  for (int o = 16; o > 0; o >>= 1) sum += __shfl_xor(sum, o, 32);
  if ((t & 31) == 0) reds[t >> 5] = sum;
  __syncthreads();
  sum = reds[0] + reds[1] + reds[2] + reds[3];
  float invs = 1.0f / sum;
  __syncthreads();

  if (t < 32) {
    float acc = 0.f;
    for (int k = 0; k < Lk; k++)
      acc += sc[k] * V[((size_t)(b * Lk + k)) * HIDDEN + h * 32 + t];
    float r = acc * invs;
    size_t oi = ((size_t)(b * Lq + q)) * HIDDEN + h * 32 + t;
    out[oi] = r;
    if (outb) outb[oi] = f2bf(r);
  }
}

// ---------------- embedding / row-gather ----------------
__global__ void gather_embed_kernel(const int* __restrict__ target, const float* __restrict__ wemb,
                                    float* __restrict__ ie)
{
  int r = blockIdx.x, t = threadIdx.x;                        // r over 200
  ie[(size_t)r * HIDDEN + t] = wemb[(size_t)target[r] * HIDDEN + t];
}
__global__ void gather_rows_kernel(const float* __restrict__ x, float* __restrict__ xr,
                                   unsigned short* __restrict__ xrb, int pos)
{
  int b = blockIdx.x, t = threadIdx.x;
  float v = x[((size_t)b * TTOK + pos) * HIDDEN + t];
  xr[(size_t)b * HIDDEN + t] = v;
  xrb[(size_t)b * HIDDEN + t] = f2bf(v);
}

// ---------------- sampling step (1 block per batch) ----------------
__global__ __launch_bounds__(256)
void sample_kernel(const float* __restrict__ logits, const float* __restrict__ wemb,
                   float* __restrict__ ie, int* __restrict__ tm, int* __restrict__ fin, int i)
{
  __shared__ float svals[256];
  __shared__ int   sidx[256];
  __shared__ float topv[5];
  __shared__ int   topi[5];
  __shared__ float probs[5];
  int b = blockIdx.x, t = threadIdx.x;
  const float* lg = logits + (size_t)b * VOCABN;
  int finb = fin[b];

  for (int pass = 0; pass < 5; pass++) {
    float bv = -3.0e38f; int bi = 0;
    for (int k = t; k < VOCABN; k += 256) {
      float v = lg[k];
      bool skip = false;
      for (int pp = 0; pp < pass; pp++) if (topi[pp] == k) skip = true;
      if (!skip && v > bv) { bv = v; bi = k; }
    }
    svals[t] = bv; sidx[t] = bi;
    __syncthreads();
    for (int sft = 128; sft > 0; sft >>= 1) {
      if (t < sft && svals[t + sft] > svals[t]) { svals[t] = svals[t + sft]; sidx[t] = sidx[t + sft]; }
      __syncthreads();
    }
    if (t == 0) { topv[pass] = svals[0]; topi[pass] = sidx[0]; }
    __syncthreads();
  }
  if (t == 0) {
    float m = topv[0];
    for (int k = 1; k < 5; k++) m = fmaxf(m, topv[k]);
    float s = 0.f;
    for (int k = 0; k < 5; k++) { probs[k] = __expf(topv[k] - m); s += probs[k]; }
    for (int k = 0; k < 5; k++) probs[k] /= s;
  }
  __syncthreads();

  int max_id = topi[0];
  int prev = (finb > 0);
  int hit  = (max_id == ENDID);
  float e;
  if (prev)      e = wemb[0 * HIDDEN + t];        // PAD_ID = 0
  else if (hit)  e = wemb[(size_t)ENDID * HIDDEN + t];
  else {
    e = 0.f;
    for (int k = 0; k < 5; k++) e += probs[k] * wemb[(size_t)topi[k] * HIDDEN + t];
  }
  ie[((size_t)b * TTOK + i) * HIDDEN + t] = e;
  if (t == 0) {
    tm[b * TTOK + i] = prev;
    if (!prev && hit) fin[b] = i - 1;
  }
}

// ---------------- output packing / finalize ----------------
__global__ void copy_carry_kernel(const float* __restrict__ logits_full, float* __restrict__ carry)
{
  int j = blockIdx.x * 256 + threadIdx.x;        // 8*30522
  if (j >= BSZ * VOCABN) return;
  int b = j / VOCABN, k = j % VOCABN;
  carry[j] = logits_full[((size_t)b * TTOK + (MAXTGT - 1)) * VOCABN + k];
}
__global__ void pack_output_kernel(const float* __restrict__ logits_full, float* __restrict__ dout)
{
  int j = blockIdx.x * 256 + threadIdx.x;        // 8*24*30522
  if (j >= BSZ * MAXTGT * VOCABN) return;
  int k  = j % VOCABN;
  int tt = (j / VOCABN) % MAXTGT;
  int b  = j / (VOCABN * MAXTGT);
  dout[j] = logits_full[((size_t)b * TTOK + tt) * VOCABN + k];
}
__global__ void pack_hs_kernel(const float* __restrict__ x3, float* __restrict__ dout)
{
  int j = blockIdx.x * 256 + threadIdx.x;        // 24*8*256, layout [t][b][d]
  if (j >= MAXTGT * BSZ * HIDDEN) return;
  int c  = j % HIDDEN;
  int b  = (j / HIDDEN) % BSZ;
  int tt = j / (HIDDEN * BSZ);
  dout[j] = x3[((size_t)b * TTOK + tt) * HIDDEN + c];
}
__global__ __launch_bounds__(256)
void finalize_kernel(const float* __restrict__ carry, int* __restrict__ tm, int* __restrict__ fin,
                     float* __restrict__ out_tm, float* __restrict__ out_fin)
{
  __shared__ float svals[256];
  __shared__ int   sidx[256];
  int b = blockIdx.x, t = threadIdx.x;
  const float* lg = carry + (size_t)b * VOCABN;
  float bv = -3.0e38f; int bi = 0;
  for (int k = t; k < VOCABN; k += 256) { float v = lg[k]; if (v > bv) { bv = v; bi = k; } }
  svals[t] = bv; sidx[t] = bi;
  __syncthreads();
  for (int sft = 128; sft > 0; sft >>= 1) {
    if (t < sft && svals[t + sft] > svals[t]) { svals[t] = svals[t + sft]; sidx[t] = sidx[t + sft]; }
    __syncthreads();
  }
  if (t == 0) {
    int unfin = (fin[b] == 0);
    if (unfin) tm[b * TTOK + MAXTGT] = 0;
    if (unfin && sidx[0] == ENDID) fin[b] = MAXTGT - 1;
  }
  __syncthreads();
  if (t < MAXTGT) out_tm[b * MAXTGT + t] = (float)tm[b * TTOK + 1 + t];
  if (t == 32)    out_fin[b] = (float)fin[b];
}

// =====================================================================================
extern "C" void kernel_launch(void* const* d_in, const int* in_sizes, int n_in,
                              void* d_out, int out_size, void* d_ws, size_t ws_size,
                              hipStream_t stream)
{
  (void)in_sizes; (void)n_in; (void)out_size; (void)ws_size;

  // ---- inputs (setup_inputs dict order, params flattened in insertion order) ----
  const float*         src      = (const float*)d_in[0];
  const unsigned char* maskb    = (const unsigned char*)d_in[1];
  const float*         pos      = (const float*)d_in[2];
  const int*           target   = (const int*)d_in[3];
  const unsigned char* tgtmaskb = (const unsigned char*)d_in[4];
  const float* proj_w  = (const float*)d_in[5];
  const float* proj_b  = (const float*)d_in[6];
  const float* wemb    = (const float*)d_in[7];
  const float* tgt_pos = (const float*)d_in[8];
  const float* emb_s   = (const float*)d_in[9];
  const float* emb_b   = (const float*)d_in[10];
  const float* esa[8]; const float* dsa[8]; const float* dca[8];
  for (int j = 0; j < 8; j++) { esa[j] = (const float*)d_in[11 + j];
                                dsa[j] = (const float*)d_in[19 + j];
                                dca[j] = (const float*)d_in[27 + j]; }
  const float* effn_w1 = (const float*)d_in[35]; const float* effn_b1 = (const float*)d_in[36];
  const float* effn_w2 = (const float*)d_in[37]; const float* effn_b2 = (const float*)d_in[38];
  const float* dffn_w1 = (const float*)d_in[39]; const float* dffn_b1 = (const float*)d_in[40];
  const float* dffn_w2 = (const float*)d_in[41]; const float* dffn_b2 = (const float*)d_in[42];
  const float* eln1_s = (const float*)d_in[43]; const float* eln1_b = (const float*)d_in[44];
  const float* eln2_s = (const float*)d_in[45]; const float* eln2_b = (const float*)d_in[46];
  const float* dln1_s = (const float*)d_in[47]; const float* dln1_b = (const float*)d_in[48];
  const float* dln2_s = (const float*)d_in[49]; const float* dln2_b = (const float*)d_in[50];
  const float* dln3_s = (const float*)d_in[51]; const float* dln3_b = (const float*)d_in[52];
  const float* mlp_w1 = (const float*)d_in[53]; const float* mlp_b1 = (const float*)d_in[54];
  const float* mlp_w2 = (const float*)d_in[55]; const float* mlp_b2 = (const float*)d_in[56];
  const float* mlp_w3 = (const float*)d_in[57]; const float* mlp_b3 = (const float*)d_in[58];

  // ---- workspace arena ----
  char* wsb = (char*)d_ws;
  size_t off = 0;
  auto alloc = [&](size_t bytes) -> void* {
    void* p = wsb + off;
    off = (off + bytes + 255) & ~(size_t)255;
    return p;
  };
  auto fb = [&](size_t elems) { return (float*)alloc(elems * 4); };
  auto ub = [&](size_t elems) { return (unsigned short*)alloc(elems * 2); };
  auto ib = [&](size_t elems) { return (int*)alloc(elems * 4); };

  // bf16 weights (transposed to [N][K] where needed)
  unsigned short* proj_wt = ub((size_t)HIDDEN * CBCH);                 // already N-major
  unsigned short* esaT[4], *dsaT[4], *dcaT[4];                         // qw kw vw ow transposed
  for (int j = 0; j < 4; j++) { esaT[j] = ub(HIDDEN * HIDDEN);
                                dsaT[j] = ub(HIDDEN * HIDDEN);
                                dcaT[j] = ub(HIDDEN * HIDDEN); }
  unsigned short* effn_w1t = ub((size_t)DFFN * HIDDEN);
  unsigned short* effn_w2t = ub((size_t)HIDDEN * DFFN);
  unsigned short* dffn_w1t = ub((size_t)DFFN * HIDDEN);
  unsigned short* dffn_w2t = ub((size_t)HIDDEN * DFFN);
  unsigned short* mlp_w1t  = ub((size_t)512 * HIDDEN);
  unsigned short* mlp_w2t  = ub((size_t)512 * 512);
  unsigned short* mlp_w3t  = ub((size_t)VOCABN * 512);

  // bf16 activations (src transpose + producer shadows)
  unsigned short* srcTb    = ub((size_t)MENC * CBCH);
  unsigned short* enc_x_b  = ub((size_t)MENC * HIDDEN);
  unsigned short* e_qk_b   = ub((size_t)MENC * HIDDEN);
  unsigned short* e_ao_b   = ub((size_t)MENC * HIDDEN);
  unsigned short* e_x1_b   = ub((size_t)MENC * HIDDEN);
  unsigned short* memory_b = ub((size_t)MENC * HIDDEN);
  unsigned short* mempos_b = ub((size_t)MENC * HIDDEN);
  unsigned short* e_ffn_b  = ub((size_t)MENC * DFFN);
  unsigned short* tgt0_b   = ub((size_t)MDEC * HIDDEN);
  unsigned short* dqk_b    = ub((size_t)MDEC * HIDDEN);
  unsigned short* dao_b    = ub((size_t)MDEC * HIDDEN);
  unsigned short* dx2_b    = ub((size_t)MDEC * HIDDEN);
  unsigned short* dx3_b    = ub((size_t)MDEC * HIDDEN);
  unsigned short* dffn_b16 = ub((size_t)MDEC * DFFN);
  unsigned short* xr_b     = ub((size_t)BSZ * HIDDEN);
  unsigned short* m1s_b    = ub((size_t)BSZ * 512);
  unsigned short* m2s_b    = ub((size_t)BSZ * 512);
  unsigned short* m1f_b    = ub((size_t)MDEC * 512);
  unsigned short* m2f_b    = ub((size_t)MDEC * 512);
  unsigned short* warm_sink= ub(1024);

  // fp32 activations
  float* posT   = fb((size_t)MENC * HIDDEN);
  float* enc_x  = fb((size_t)MENC * HIDDEN);
  float* e_qk   = fb((size_t)MENC * HIDDEN);
  float* e_q    = fb((size_t)MENC * HIDDEN);
  float* e_k    = fb((size_t)MENC * HIDDEN);
  float* e_v    = fb((size_t)MENC * HIDDEN);
  float* e_ao   = fb((size_t)MENC * HIDDEN);
  float* e_tmp  = fb((size_t)MENC * HIDDEN);
  float* e_x1   = fb((size_t)MENC * HIDDEN);
  float* memory = fb((size_t)MENC * HIDDEN);
  float* mempos = fb((size_t)MENC * HIDDEN);
  float* Kmem   = fb((size_t)MENC * HIDDEN);
  float* Vmem   = fb((size_t)MENC * HIDDEN);
  float* e_ffn  = fb((size_t)MENC * DFFN);
  int*   kpmem  = ib(MENC);

  float* ie    = fb((size_t)MDEC * HIDDEN);
  float* te    = fb((size_t)MDEC * HIDDEN);
  float* tgt0  = fb((size_t)MDEC * HIDDEN);
  float* dqk   = fb((size_t)MDEC * HIDDEN);
  float* dq    = fb((size_t)MDEC * HIDDEN);
  float* dk    = fb((size_t)MDEC * HIDDEN);
  float* dv    = fb((size_t)MDEC * HIDDEN);
  float* dao   = fb((size_t)MDEC * HIDDEN);
  float* dtmp  = fb((size_t)MDEC * HIDDEN);
  float* dx1   = fb((size_t)MDEC * HIDDEN);
  float* dx2   = fb((size_t)MDEC * HIDDEN);
  float* dx3   = fb((size_t)MDEC * HIDDEN);
  float* d_ffn = fb((size_t)MDEC * DFFN);
  float* xr    = fb((size_t)BSZ * HIDDEN);
  float* m1s   = fb((size_t)BSZ * 512);
  float* m2s   = fb((size_t)BSZ * 512);
  float* m1f   = fb((size_t)MDEC * 512);
  float* m2f   = fb((size_t)MDEC * 512);
  float* logits_full = fb((size_t)MDEC * VOCABN);
  float* carry = fb((size_t)BSZ * VOCABN);
  int*   tm    = ib(BSZ * TTOK);
  int*   fin   = ib(BSZ);

  auto G = [](int n) { return (n + 255) / 256; };
  auto gemmB = [&](const unsigned short* Abf, const unsigned short* Bt, const float* bias,
                   float* C, unsigned short* Cb, int M, int N, int K, int relu) {
    int tilesN4 = (N + 63) / 64;
    int tiles   = ((M + 15) / 16) * tilesN4;
    gemm_wmma_kernel<<<(tiles + 7) / 8, 256, 0, stream>>>(Abf, Bt, bias, C, Cb,
                                                          M, N, K, relu, tilesN4, tiles);
  };
  auto ln = [&](const float* x, const float* res, const float* s, const float* b,
                float* o, unsigned short* ob, int rows) {
    ln_kernel<<<(rows + 7) / 8, 256, 0, stream>>>(x, res, s, b, o, ob, rows);
  };

  // ---- weight conversion (idempotent, per launch) ----
  conv_bf16_kernel<<<G(HIDDEN * CBCH), 256, 0, stream>>>(proj_w, proj_wt, HIDDEN * CBCH);
  for (int j = 0; j < 4; j++) {
    convt_bf16_kernel<<<G(HIDDEN * HIDDEN), 256, 0, stream>>>(esa[2 * j], esaT[j], HIDDEN, HIDDEN);
    convt_bf16_kernel<<<G(HIDDEN * HIDDEN), 256, 0, stream>>>(dsa[2 * j], dsaT[j], HIDDEN, HIDDEN);
    convt_bf16_kernel<<<G(HIDDEN * HIDDEN), 256, 0, stream>>>(dca[2 * j], dcaT[j], HIDDEN, HIDDEN);
  }
  convt_bf16_kernel<<<G(HIDDEN * DFFN), 256, 0, stream>>>(effn_w1, effn_w1t, HIDDEN, DFFN);
  convt_bf16_kernel<<<G(HIDDEN * DFFN), 256, 0, stream>>>(effn_w2, effn_w2t, DFFN, HIDDEN);
  convt_bf16_kernel<<<G(HIDDEN * DFFN), 256, 0, stream>>>(dffn_w1, dffn_w1t, HIDDEN, DFFN);
  convt_bf16_kernel<<<G(HIDDEN * DFFN), 256, 0, stream>>>(dffn_w2, dffn_w2t, DFFN, HIDDEN);
  convt_bf16_kernel<<<G(HIDDEN * 512), 256, 0, stream>>>(mlp_w1, mlp_w1t, HIDDEN, 512);
  convt_bf16_kernel<<<G(512 * 512), 256, 0, stream>>>(mlp_w2, mlp_w2t, 512, 512);
  convt_bf16_kernel<<<G(512 * VOCABN), 256, 0, stream>>>(mlp_w3, mlp_w3t, 512, VOCABN);

  // TDM: pull the 31MB vocab-head weight through L2 before the serial decode chain
  warm_l2_kernel<<<(VOCABN * 512 + WARM_CHUNK - 1) / WARM_CHUNK, 32, 0, stream>>>(mlp_w3t, warm_sink);

  // ---- input prep ----
  transpose_src_kernel<<<G(MENC * CBCH), 256, 0, stream>>>(src, srcTb);
  transpose_pos_kernel<<<G(MENC * HIDDEN), 256, 0, stream>>>(pos, posT);
  bool_to_int_kernel<<<G(MENC), 256, 0, stream>>>(maskb, kpmem, MENC);
  bool_to_int_kernel<<<G(MDEC), 256, 0, stream>>>(tgtmaskb, tm, MDEC);
  fill_int_kernel<<<1, 256, 0, stream>>>(fin, 0, BSZ);

  // ---- encoder ----
  gemmB(srcTb, proj_wt, proj_b, enc_x, enc_x_b, MENC, HIDDEN, CBCH, 0);      // src_tok
  add_kernel<<<G(MENC * HIDDEN), 256, 0, stream>>>(enc_x, posT, e_qk, e_qk_b, MENC * HIDDEN);
  gemmB(e_qk_b,  esaT[0], esa[1], e_q, nullptr, MENC, HIDDEN, HIDDEN, 0);
  gemmB(e_qk_b,  esaT[1], esa[3], e_k, nullptr, MENC, HIDDEN, HIDDEN, 0);
  gemmB(enc_x_b, esaT[2], esa[5], e_v, nullptr, MENC, HIDDEN, HIDDEN, 0);
  attn_kernel<<<BSZ * NHEADS * HW2, 128, 0, stream>>>(e_q, e_k, e_v, e_ao, e_ao_b, HW2, HW2, 0, kpmem);
  gemmB(e_ao_b, esaT[3], esa[7], e_tmp, nullptr, MENC, HIDDEN, HIDDEN, 0);
  ln(e_tmp, enc_x, eln1_s, eln1_b, e_x1, e_x1_b, MENC);
  gemmB(e_x1_b, effn_w1t, effn_b1, e_ffn, e_ffn_b, MENC, DFFN, HIDDEN, 1);
  gemmB(e_ffn_b, effn_w2t, effn_b2, e_tmp, nullptr, MENC, HIDDEN, DFFN, 0);
  ln(e_tmp, e_x1, eln2_s, eln2_b, memory, memory_b, MENC);

  // cross-attn K/V are fixed for the whole decode loop
  add_kernel<<<G(MENC * HIDDEN), 256, 0, stream>>>(memory, posT, mempos, mempos_b, MENC * HIDDEN);
  gemmB(mempos_b, dcaT[1], dca[3], Kmem, nullptr, MENC, HIDDEN, HIDDEN, 0);
  gemmB(memory_b, dcaT[2], dca[5], Vmem, nullptr, MENC, HIDDEN, HIDDEN, 0);

  // initial embeddings
  gather_embed_kernel<<<MDEC, HIDDEN, 0, stream>>>(target, wemb, ie);

  auto decode = [&](int head_pos /* >=0: single position head; -1: full head */) {
    add_rows_kernel<<<G(MDEC * HIDDEN), 256, 0, stream>>>(ie, tgt_pos, te, nullptr, MDEC, TTOK);
    ln(te, nullptr, emb_s, emb_b, tgt0, tgt0_b, MDEC);
    // self-attention (causal + tm key-padding)
    add_rows_kernel<<<G(MDEC * HIDDEN), 256, 0, stream>>>(tgt0, tgt_pos, dqk, dqk_b, MDEC, TTOK);
    gemmB(dqk_b,  dsaT[0], dsa[1], dq, nullptr, MDEC, HIDDEN, HIDDEN, 0);
    gemmB(dqk_b,  dsaT[1], dsa[3], dk, nullptr, MDEC, HIDDEN, HIDDEN, 0);
    gemmB(tgt0_b, dsaT[2], dsa[5], dv, nullptr, MDEC, HIDDEN, HIDDEN, 0);
    attn_kernel<<<BSZ * NHEADS * TTOK, 128, 0, stream>>>(dq, dk, dv, dao, dao_b, TTOK, TTOK, 1, tm);
    gemmB(dao_b, dsaT[3], dsa[7], dtmp, nullptr, MDEC, HIDDEN, HIDDEN, 0);
    ln(dtmp, tgt0, dln1_s, dln1_b, dx1, nullptr, MDEC);
    // cross-attention
    add_rows_kernel<<<G(MDEC * HIDDEN), 256, 0, stream>>>(dx1, tgt_pos, dqk, dqk_b, MDEC, TTOK);
    gemmB(dqk_b, dcaT[0], dca[1], dq, nullptr, MDEC, HIDDEN, HIDDEN, 0);
    attn_kernel<<<BSZ * NHEADS * TTOK, 128, 0, stream>>>(dq, Kmem, Vmem, dao, dao_b, TTOK, HW2, 0, kpmem);
    gemmB(dao_b, dcaT[3], dca[7], dtmp, nullptr, MDEC, HIDDEN, HIDDEN, 0);
    ln(dtmp, dx1, dln2_s, dln2_b, dx2, dx2_b, MDEC);
    // FFN
    gemmB(dx2_b, dffn_w1t, dffn_b1, d_ffn, dffn_b16, MDEC, DFFN, HIDDEN, 1);
    gemmB(dffn_b16, dffn_w2t, dffn_b2, dtmp, nullptr, MDEC, HIDDEN, DFFN, 0);
    ln(dtmp, dx2, dln3_s, dln3_b, dx3, dx3_b, MDEC);
    // vocab head
    if (head_pos >= 0) {
      gather_rows_kernel<<<BSZ, HIDDEN, 0, stream>>>(dx3, xr, xr_b, head_pos);
      gemmB(xr_b,  mlp_w1t, mlp_b1, m1s, m1s_b, BSZ, 512, HIDDEN, 1);
      gemmB(m1s_b, mlp_w2t, mlp_b2, m2s, m2s_b, BSZ, 512, 512, 1);
      gemmB(m2s_b, mlp_w3t, mlp_b3, carry, nullptr, BSZ, VOCABN, 512, 0);
    } else {
      gemmB(dx3_b, mlp_w1t, mlp_b1, m1f, m1f_b, MDEC, 512, HIDDEN, 1);
      gemmB(m1f_b, mlp_w2t, mlp_b2, m2f, m2f_b, MDEC, 512, 512, 1);
      gemmB(m2f_b, mlp_w3t, mlp_b3, logits_full, nullptr, MDEC, VOCABN, 512, 0);
    }
  };

  // ---- decode loop: initial + 23 scan steps (last step gets the full head) ----
  decode(0);                                                  // out = output[:, 0]
  for (int i = 1; i < MAXTGT - 1; i++) {
    sample_kernel<<<BSZ, 256, 0, stream>>>(carry, wemb, ie, tm, fin, i);
    decode(i);                                                // out = output[:, i]
  }
  sample_kernel<<<BSZ, 256, 0, stream>>>(carry, wemb, ie, tm, fin, MAXTGT - 1);
  decode(-1);                                                 // final: full logits + hs
  copy_carry_kernel<<<G(BSZ * VOCABN), 256, 0, stream>>>(logits_full, carry);

  // ---- pack outputs: [memory | output[:, :24] | hs_t[:24] | tm[:,1:25] | fin] ----
  float* dout = (float*)d_out;
  const size_t MEM_OFF = 0;
  const size_t OUT_OFF = MEM_OFF + (size_t)BSZ * HW2 * HIDDEN;       // 819200
  const size_t HS_OFF  = OUT_OFF + (size_t)BSZ * MAXTGT * VOCABN;    // +5860224
  const size_t TM_OFF  = HS_OFF + (size_t)MAXTGT * BSZ * HIDDEN;     // +49152
  const size_t FIN_OFF = TM_OFF + (size_t)BSZ * MAXTGT;              // +192

  finalize_kernel<<<BSZ, 256, 0, stream>>>(carry, tm, fin, dout + TM_OFF, dout + FIN_OFF);
  copy_kernel<<<G(MENC * HIDDEN), 256, 0, stream>>>(memory, dout + MEM_OFF, MENC * HIDDEN);
  pack_output_kernel<<<G(BSZ * MAXTGT * VOCABN), 256, 0, stream>>>(logits_full, dout + OUT_OFF);
  pack_hs_kernel<<<G(MAXTGT * BSZ * HIDDEN), 256, 0, stream>>>(dx3, dout + HS_OFF);
}
